// ASGFormer_68564857913542
// MI455X (gfx1250) — compile-verified, matrix operand order checked by reading, not run on hardware
//
#include <hip/hip_runtime.h>
#include <hip/hip_bf16.h>

// ---------------------------------------------------------------------------
// Types for CDNA5 WMMA (wave32, v_wmma_f32_16x16x32_bf16, f32 accumulate)
// ---------------------------------------------------------------------------
typedef __bf16 v16bf __attribute__((ext_vector_type(16)));
typedef float  v8f   __attribute__((ext_vector_type(8)));

union Frag16 {
    v16bf          v;
    unsigned short u[16];
    uint4          q[2];
};

__device__ __forceinline__ unsigned short f2bf(float f) {
    __hip_bfloat16 h = __float2bfloat16(f);
    unsigned short u;
    __builtin_memcpy(&u, &h, sizeof(u));
    return u;
}

// Fragment layout (ISA 7.12.2), identical for A (rows on lanes) and B (cols on
// lanes): lane half h = lane>>4, index r = lane&15 selects row (A) / col (B);
// the 16 halves per lane are two contiguous 8-element K-chunks at
// kb + h*8 and kb + 16 + h*8.  With A row-major [M][K] and W pre-transposed to
// [Nc][K], BOTH fragments load as two b128s.
__device__ __forceinline__ void load_frag(Frag16& f, const unsigned short* row_base,
                                          int kb, int half) {
    const unsigned short* p = row_base + kb + half * 8;
    f.q[0] = *(const uint4*)(p);
    f.q[1] = *(const uint4*)(p + 16);
}

__device__ __forceinline__ v8f wmma_bf16(const Frag16& a, const Frag16& b, v8f c) {
    return __builtin_amdgcn_wmma_f32_16x16x32_bf16(false, a.v, false, b.v,
                                                   (short)0, c, false, false);
}

// ---------------------------------------------------------------------------
// fp32 -> bf16 conversion (row-major copy)
// ---------------------------------------------------------------------------
__global__ __launch_bounds__(256) void cvt_bf16_kernel(const float* __restrict__ in,
                                                       unsigned short* __restrict__ out,
                                                       int n) {
    int t = blockIdx.x * 256 + threadIdx.x;
    if (t < n) out[t] = f2bf(in[t]);
}

// fp32 W[K][Nc] -> bf16 Wt[Nc][K] (transpose for b128 fragment loads)
__global__ __launch_bounds__(256) void cvt_transpose_kernel(
    const float* __restrict__ in, unsigned short* __restrict__ out, int K, int Nc) {
    int t = blockIdx.x * 256 + threadIdx.x;
    if (t >= K * Nc) return;
    const int n = t / K;
    const int k = t - n * K;
    out[t] = f2bf(in[(size_t)k * Nc + n]);
}

// ---------------------------------------------------------------------------
// bf16 WMMA GEMM: C[M,Nc] = A[M,K] @ W[K,Nc] + bias (W given transposed
// [Nc][K]).  Block = 256 threads (8 waves).  The 16-row A tile is staged once
// in LDS; each wave computes two 16x16 column tiles (32 cols), so a block
// covers 256 columns: grid = (M/16, Nc/256).  K % 32 == 0, K <= 1024.
// ---------------------------------------------------------------------------
template <int RELU, int OUTF, int OUTB>
__global__ __launch_bounds__(256) void gemm_bf16_kernel(
    const unsigned short* __restrict__ A, const unsigned short* __restrict__ Wt,
    const float* __restrict__ bias, float* __restrict__ Cf,
    unsigned short* __restrict__ Cb, int M, int K, int Nc) {

    __shared__ unsigned short sA[16 * 1024];  // 32 KB (max K = 1024)

    const int tid = threadIdx.x;
    const int m0  = blockIdx.x * 16;

    // Cooperative copy of the 16xK bf16 A tile into LDS (b128 granularity).
    {
        const uint4* src = (const uint4*)(A + (size_t)m0 * K);
        uint4*       dst = (uint4*)sA;
        const int totalq = 16 * K / 8;  // uint4 = 8 halves
        for (int i = tid; i < totalq; i += 256) dst[i] = src[i];
    }
    __syncthreads();

    const int wave = tid >> 5;
    const int lane = tid & 31;
    const int half = lane >> 4;
    const int r    = lane & 15;
    const int cb   = blockIdx.y * 256 + wave * 32;
    const int c0   = cb + r;
    const int c1   = cb + 16 + r;

    const unsigned short* arow = &sA[r * K];
    const unsigned short* w0   = Wt + (size_t)c0 * K;
    const unsigned short* w1   = Wt + (size_t)c1 * K;

    v8f acc0 = {0.f, 0.f, 0.f, 0.f, 0.f, 0.f, 0.f, 0.f};
    v8f acc1 = acc0;
    for (int kb = 0; kb < K; kb += 32) {
        Frag16 a, b0, b1;
        load_frag(a, arow, kb, half);
        load_frag(b0, w0, kb, half);
        load_frag(b1, w1, kb, half);
        acc0 = wmma_bf16(a, b0, acc0);
        acc1 = wmma_bf16(a, b1, acc1);
    }

    const float bc0 = bias[c0];
    const float bc1 = bias[c1];
#pragma unroll
    for (int j = 0; j < 8; ++j) {
        const size_t row = m0 + j + half * 8;
        float v0 = acc0[j] + bc0;
        float v1 = acc1[j] + bc1;
        if (RELU) { v0 = fmaxf(v0, 0.f); v1 = fmaxf(v1, 0.f); }
        if (OUTF) {
            Cf[row * Nc + c0] = v0;
            Cf[row * Nc + c1] = v1;
        }
        if (OUTB) {
            Cb[row * Nc + c0] = f2bf(v0);
            Cb[row * Nc + c1] = f2bf(v1);
        }
    }
}

// ---------------------------------------------------------------------------
// Fused positional-embedding + attention-score kernel, 16 edges per block.
//   h  = relu(dp @ Wp1 + bp1)            (VALU, K=3)
//   pe = LayerNorm(h @ Wp2 + bp2)        (WMMA over LDS-staged bf16 tile)
//   ke = pe + Knode[src]
//   score[e,h] = (Qnode[dst,h,:] . ke[h,:]) / sqrt(32)
//   atomicMax(per-(dst,head) max, order-preserving uint encoding)
// ---------------------------------------------------------------------------
__global__ __launch_bounds__(256) void pe_score_kernel(
    const float* __restrict__ pos, const int* __restrict__ ei,
    const float* __restrict__ Wp1, const float* __restrict__ bp1,
    const unsigned short* __restrict__ Wp2t,  // [256][256] transposed bf16
    const float* __restrict__ bp2,
    const float* __restrict__ gp, const float* __restrict__ bpn,
    const float* __restrict__ Qn, const float* __restrict__ Kn,
    float* __restrict__ scoreE, unsigned* __restrict__ smaxU, int E) {

    __shared__ unsigned short hA[16 * 256];  // bf16 A tile (16 edges x 256)
    __shared__ float peT[16 * 256];          // fp32 pe tile -> (pe + K[src])
    __shared__ int   s_src[16], s_dst[16];
    __shared__ float s_dp[16][3];

    const int tid = threadIdx.x;
    const int e0  = blockIdx.x * 16;

    if (tid < 16) {
        const int e = e0 + tid;
        const int s = ei[e];       // edge_index[0]: source (sender)
        const int d = ei[E + e];   // edge_index[1]: target (segment)
        s_src[tid] = s;
        s_dst[tid] = d;
#pragma unroll
        for (int c = 0; c < 3; ++c) s_dp[tid][c] = pos[d * 3 + c] - pos[s * 3 + c];
    }
    __syncthreads();

    // h = relu(dp @ Wp1 + bp1), stored bf16 in LDS. 16 cols per thread.
    {
        const int row = tid & 15;
        const int c0  = (tid >> 4) * 16;
        const float d0 = s_dp[row][0], d1 = s_dp[row][1], d2 = s_dp[row][2];
#pragma unroll
        for (int c = c0; c < c0 + 16; ++c) {
            float hv = d0 * Wp1[c] + d1 * Wp1[256 + c] + d2 * Wp1[512 + c] + bp1[c];
            hA[row * 256 + c] = f2bf(fmaxf(hv, 0.f));
        }
    }
    __syncthreads();

    // WMMA: wave w computes output columns [w*32, w*32+32)
    const int wave = tid >> 5;
    const int lane = tid & 31;
    const int half = lane >> 4;
    const int r    = lane & 15;
    const int n0   = wave * 32;

    const unsigned short* arow = &hA[r * 256];
    const unsigned short* w0   = Wp2t + (size_t)(n0 + r) * 256;
    const unsigned short* w1   = Wp2t + (size_t)(n0 + 16 + r) * 256;

    v8f acc0 = {0.f, 0.f, 0.f, 0.f, 0.f, 0.f, 0.f, 0.f};
    v8f acc1 = acc0;
#pragma unroll
    for (int kb = 0; kb < 256; kb += 32) {
        Frag16 a, b0, b1;
        load_frag(a, arow, kb, half);
        load_frag(b0, w0, kb, half);
        load_frag(b1, w1, kb, half);
        acc0 = wmma_bf16(a, b0, acc0);
        acc1 = wmma_bf16(a, b1, acc1);
    }
    const float bc0 = bp2[n0 + r], bc1 = bp2[n0 + 16 + r];
#pragma unroll
    for (int j = 0; j < 8; ++j) {
        const int row = j + half * 8;
        peT[row * 256 + n0 + r]      = acc0[j] + bc0;
        peT[row * 256 + n0 + 16 + r] = acc1[j] + bc1;
    }
    __syncthreads();

    // LayerNorm per edge row + add Knode[src]. Wave w handles rows 2w, 2w+1.
#pragma unroll
    for (int rr = 0; rr < 2; ++rr) {
        const int row = wave * 2 + rr;
        float s = 0.f, s2 = 0.f;
#pragma unroll
        for (int i = 0; i < 8; ++i) {
            const float v = peT[row * 256 + lane + 32 * i];
            s += v; s2 += v * v;
        }
#pragma unroll
        for (int off = 16; off; off >>= 1) {
            s  += __shfl_xor(s, off, 32);
            s2 += __shfl_xor(s2, off, 32);
        }
        const float mean = s * (1.f / 256.f);
        const float var  = s2 * (1.f / 256.f) - mean * mean;
        const float rs   = rsqrtf(var + 1e-5f);
        const int   srcn = s_src[row];
#pragma unroll
        for (int i = 0; i < 8; ++i) {
            const int c = lane + 32 * i;
            const float v = (peT[row * 256 + c] - mean) * rs * gp[c] + bpn[c];
            peT[row * 256 + c] = v + Kn[(size_t)srcn * 256 + c];
        }
    }
    __syncthreads();

    // Scores: 16 edges x 8 heads = 128 (edge, head) pairs.
    if (tid < 128) {
        const int e  = tid >> 3;
        const int hh = tid & 7;
        const int dn = s_dst[e];
        const float* qp = &Qn[(size_t)dn * 256 + hh * 32];
        const float* kp = &peT[e * 256 + hh * 32];
        float sc = 0.f;
#pragma unroll
        for (int j = 0; j < 32; ++j) sc += qp[j] * kp[j];
        sc *= 0.17677669529663687f;  // 1/sqrt(32)
        scoreE[(size_t)(e0 + e) * 8 + hh] = sc;
        const unsigned bits = __float_as_uint(sc);
        const unsigned key  = bits ^ (((int)bits >> 31) | 0x80000000u);
        atomicMax(&smaxU[dn * 8 + hh], key);
    }
}

// ---------------------------------------------------------------------------
// score -> exp(score - segment_max); atomicAdd into segment sum.
// ---------------------------------------------------------------------------
__global__ __launch_bounds__(256) void softmax_num_kernel(
    const int* __restrict__ ei, float* __restrict__ scoreE,
    const unsigned* __restrict__ smaxU, float* __restrict__ ssum, int E) {
    const int t = blockIdx.x * 256 + threadIdx.x;
    if (t >= E * 8) return;
    const int e = t >> 3, hh = t & 7;
    const int dn = ei[E + e];
    const unsigned key = smaxU[dn * 8 + hh];
    const unsigned bits = (key & 0x80000000u) ? (key ^ 0x80000000u) : ~key;
    const float mx = __uint_as_float(bits);
    const float sc = __expf(scoreE[t] - mx);
    scoreE[t] = sc;
    atomicAdd(&ssum[dn * 8 + hh], sc);
}

// ---------------------------------------------------------------------------
// msg = (attn * v) scatter-add over dst. One block (256 thr) per edge.
// ---------------------------------------------------------------------------
__global__ __launch_bounds__(256) void scatter_msg_kernel(
    const int* __restrict__ ei, const float* __restrict__ scoreE,
    const float* __restrict__ ssum, const float* __restrict__ Vn,
    float* __restrict__ aggr, int E) {
    const int e = blockIdx.x;
    const int d = threadIdx.x;
    const int hh = d >> 5;
    const int sn = ei[e];
    const int dn = ei[E + e];
    const float attn = scoreE[(size_t)e * 8 + hh] / (ssum[dn * 8 + hh] + 1e-16f);
    atomicAdd(&aggr[(size_t)dn * 256 + d], attn * Vn[(size_t)sn * 256 + d]);
}

// ---------------------------------------------------------------------------
// out = LayerNorm(a + b) over 256-wide rows; optional fp32 + bf16 outputs.
// One wave per row, 8 rows per block.
// ---------------------------------------------------------------------------
__global__ __launch_bounds__(256) void add_ln_kernel(
    const float* __restrict__ a, const float* __restrict__ b,
    const float* __restrict__ g, const float* __restrict__ beta,
    float* __restrict__ outf, unsigned short* __restrict__ outbf) {
    const int wave = threadIdx.x >> 5;
    const int lane = threadIdx.x & 31;
    const size_t row = (size_t)blockIdx.x * 8 + wave;
    const float* ap = a + row * 256;
    const float* bp = b + row * 256;
    float vals[8];
    float s = 0.f, s2 = 0.f;
#pragma unroll
    for (int i = 0; i < 8; ++i) {
        const float v = ap[lane + 32 * i] + bp[lane + 32 * i];
        vals[i] = v; s += v; s2 += v * v;
    }
#pragma unroll
    for (int off = 16; off; off >>= 1) {
        s  += __shfl_xor(s, off, 32);
        s2 += __shfl_xor(s2, off, 32);
    }
    const float mean = s * (1.f / 256.f);
    const float var  = s2 * (1.f / 256.f) - mean * mean;
    const float rs   = rsqrtf(var + 1e-5f);
#pragma unroll
    for (int i = 0; i < 8; ++i) {
        const int c = lane + 32 * i;
        const float v = (vals[i] - mean) * rs * g[c] + beta[c];
        if (outf)  outf[row * 256 + c]  = v;
        if (outbf) outbf[row * 256 + c] = f2bf(v);
    }
}

// ---------------------------------------------------------------------------
// Host launch
// ---------------------------------------------------------------------------
extern "C" void kernel_launch(void* const* d_in, const int* in_sizes, int n_in,
                              void* d_out, int out_size, void* d_ws, size_t ws_size,
                              hipStream_t stream) {
    const float* x   = (const float*)d_in[0];
    const float* pos = (const float*)d_in[1];
    const int*   ei  = (const int*)d_in[2];
    const float* Wq  = (const float*)d_in[3];
    const float* bq  = (const float*)d_in[4];
    const float* Wk  = (const float*)d_in[5];
    const float* bk  = (const float*)d_in[6];
    const float* Wv  = (const float*)d_in[7];
    const float* bv  = (const float*)d_in[8];
    const float* Wp1 = (const float*)d_in[9];
    const float* bp1 = (const float*)d_in[10];
    const float* Wp2 = (const float*)d_in[11];
    const float* bp2 = (const float*)d_in[12];
    const float* gp  = (const float*)d_in[13];
    const float* bpn = (const float*)d_in[14];
    const float* Wo  = (const float*)d_in[15];
    const float* bo  = (const float*)d_in[16];
    const float* g1  = (const float*)d_in[17];
    const float* b1n = (const float*)d_in[18];
    const float* Wf1 = (const float*)d_in[19];
    const float* bf1 = (const float*)d_in[20];
    const float* Wf2 = (const float*)d_in[21];
    const float* bf2 = (const float*)d_in[22];
    const float* g2  = (const float*)d_in[23];
    const float* b2n = (const float*)d_in[24];

    const int D  = 256;
    const int Dh = 1024;
    const int N  = in_sizes[0] / D;   // 10000 (multiple of 16)
    const int E  = in_sizes[2] / 2;   // 160000 (multiple of 16)

    // Workspace carve-out (256B aligned).
    char* base = (char*)d_ws;
    size_t off = 0;
    auto carve = [&](size_t bytes) -> char* {
        char* p = base + off;
        off = (off + bytes + 255) & ~(size_t)255;
        return p;
    };
    unsigned short* xbf    = (unsigned short*)carve((size_t)N * D * 2);
    unsigned short* wqt    = (unsigned short*)carve((size_t)D * D * 2);
    unsigned short* wkt    = (unsigned short*)carve((size_t)D * D * 2);
    unsigned short* wvt    = (unsigned short*)carve((size_t)D * D * 2);
    unsigned short* wot    = (unsigned short*)carve((size_t)D * D * 2);
    unsigned short* wp2t   = (unsigned short*)carve((size_t)D * D * 2);
    unsigned short* wf1t   = (unsigned short*)carve((size_t)D * Dh * 2);
    unsigned short* wf2t   = (unsigned short*)carve((size_t)Dh * D * 2);
    float*          Qn     = (float*)carve((size_t)N * D * 4);
    float*          Kn     = (float*)carve((size_t)N * D * 4);
    float*          Vn     = (float*)carve((size_t)N * D * 4);
    float*          scoreE = (float*)carve((size_t)E * 8 * 4);
    unsigned*       smaxU  = (unsigned*)carve((size_t)N * 8 * 4);
    float*          ssum   = (float*)carve((size_t)N * 8 * 4);
    float*          aggr   = (float*)carve((size_t)N * D * 4);
    unsigned short* aggrbf = (unsigned short*)carve((size_t)N * D * 2);
    float*          outp   = (float*)carve((size_t)N * D * 4);
    float*          x1f    = (float*)carve((size_t)N * D * 4);
    unsigned short* x1bf   = (unsigned short*)carve((size_t)N * D * 2);
    unsigned short* hfbf   = (unsigned short*)carve((size_t)N * Dh * 2);
    float*          ffn    = (float*)carve((size_t)N * D * 4);
    (void)ws_size;

    // Zero accumulators (graph-capturable memset nodes).
    hipMemsetAsync(aggr, 0, (size_t)N * D * 4, stream);
    hipMemsetAsync(ssum, 0, (size_t)N * 8 * 4, stream);
    hipMemsetAsync(smaxU, 0, (size_t)N * 8 * 4, stream);  // 0 < encode(any real score)

    auto cvt = [&](const float* in, unsigned short* out, int n) {
        cvt_bf16_kernel<<<(n + 255) / 256, 256, 0, stream>>>(in, out, n);
    };
    auto cvtT = [&](const float* in, unsigned short* out, int K, int Nc) {
        cvt_transpose_kernel<<<(K * Nc + 255) / 256, 256, 0, stream>>>(in, out, K, Nc);
    };
    cvt(x, xbf, N * D);
    cvtT(Wq, wqt, D, D);
    cvtT(Wk, wkt, D, D);
    cvtT(Wv, wvt, D, D);
    cvtT(Wo, wot, D, D);
    cvtT(Wp2, wp2t, D, D);
    cvtT(Wf1, wf1t, D, Dh);
    cvtT(Wf2, wf2t, Dh, D);

    const dim3 blk(256);
    // Node-level Q/K/V GEMMs: (x @ W) then gather per-edge later.
    gemm_bf16_kernel<0, 1, 0><<<dim3(N / 16, 1), blk, 0, stream>>>(
        xbf, wqt, bq, Qn, nullptr, N, D, D);
    gemm_bf16_kernel<0, 1, 0><<<dim3(N / 16, 1), blk, 0, stream>>>(
        xbf, wkt, bk, Kn, nullptr, N, D, D);
    gemm_bf16_kernel<0, 1, 0><<<dim3(N / 16, 1), blk, 0, stream>>>(
        xbf, wvt, bv, Vn, nullptr, N, D, D);

    // Fused pe + scores + segment max.
    pe_score_kernel<<<E / 16, blk, 0, stream>>>(
        pos, ei, Wp1, bp1, wp2t, bp2, gp, bpn, Qn, Kn, scoreE, smaxU, E);

    // Softmax numerators + segment sums.
    softmax_num_kernel<<<(E * 8 + 255) / 256, blk, 0, stream>>>(
        ei, scoreE, smaxU, ssum, E);

    // Weighted message scatter-add.
    scatter_msg_kernel<<<E, blk, 0, stream>>>(ei, scoreE, ssum, Vn, aggr, E);

    // out = aggr @ Wo + bo ; x1 = LN(out + x)
    cvt(aggr, aggrbf, N * D);
    gemm_bf16_kernel<0, 1, 0><<<dim3(N / 16, 1), blk, 0, stream>>>(
        aggrbf, wot, bo, outp, nullptr, N, D, D);
    add_ln_kernel<<<N / 8, blk, 0, stream>>>(outp, x, g1, b1n, x1f, x1bf);

    // FFN: relu(x1 @ Wf1 + bf1) @ Wf2 + bf2 ; out = LN(ffn + x1)
    gemm_bf16_kernel<1, 0, 1><<<dim3(N / 16, Dh / 256), blk, 0, stream>>>(
        x1bf, wf1t, bf1, nullptr, hfbf, N, D, Dh);
    gemm_bf16_kernel<0, 1, 0><<<dim3(N / 16, 1), blk, 0, stream>>>(
        hfbf, wf2t, bf2, ffn, nullptr, N, Dh, D);
    add_ln_kernel<<<N / 8, blk, 0, stream>>>(ffn, x1f, g2, b2n, (float*)d_out, nullptr);
}
